// PodNetClassifier_6390911336757
// MI455X (gfx1250) — matrix operand order repaired and checked by reading, not compiled, not run on hardware
//
#include <hip/hip_runtime.h>

typedef __attribute__((ext_vector_type(16))) _Float16 v16h;
typedef __attribute__((ext_vector_type(8)))  _Float16 v8h;
typedef __attribute__((ext_vector_type(2)))  _Float16 v2h;
typedef __attribute__((ext_vector_type(8)))  float    v8f;

#define BATCHN   8192
#define DFEAT    64
#define KPROX    10
#define CLASSES  1000
#define THROWS   (KPROX*CLASSES)   /* 10000 */
#define NEPS     1e-12f

// ---------------------------------------------------------------------------
// Kernel 1: per-row L2 normalize (x rows and gathered theta rows) -> f16,
// plus fp32 squared norms of the normalized rows. One wave32 per row.
// ---------------------------------------------------------------------------
__global__ __launch_bounds__(256) void podnet_prep_kernel(
    const float* __restrict__ x, const float* __restrict__ theta,
    _Float16* __restrict__ aH, _Float16* __restrict__ tH,
    float* __restrict__ na2, float* __restrict__ nt2)
{
    const int gid  = blockIdx.x * blockDim.x + threadIdx.x;
    const int w    = gid >> 5;          // wave id == row id
    const int lane = gid & 31;

    float e0, e1;
    if (w < BATCHN) {
        const float* p = x + (size_t)w * DFEAT + 2 * lane;
        e0 = p[0]; e1 = p[1];
    } else if (w < BATCHN + THROWS) {
        const int r = w - BATCHN;
        const int c = r / KPROX, j = r % KPROX;   // th row r = c*K + j
        const int d0 = 2 * lane;
        // theta shape (64, 10, 1000): theta[d][j][c]
        e0 = theta[(size_t)d0       * (KPROX * CLASSES) + j * CLASSES + c];
        e1 = theta[(size_t)(d0 + 1) * (KPROX * CLASSES) + j * CLASSES + c];
    } else {
        return;
    }

    float ss = e0 * e0 + e1 * e1;
#pragma unroll
    for (int m = 16; m >= 1; m >>= 1) ss += __shfl_xor(ss, m, 32);

    const float inv = 1.0f / fmaxf(sqrtf(ss), NEPS);
    v2h hv;
    hv[0] = (_Float16)(e0 * inv);
    hv[1] = (_Float16)(e1 * inv);
    const float nrm2 = ss * inv * inv;   // |normalized row|^2 (fp32)

    if (w < BATCHN) {
        *(v2h*)(aH + (size_t)w * DFEAT + 2 * lane) = hv;
        if (lane == 0) na2[w] = nrm2;
    } else {
        const int r = w - BATCHN;
        *(v2h*)(tH + (size_t)r * DFEAT + 2 * lane) = hv;
        if (lane == 0) nt2[r] = nrm2;
    }
}

// ---------------------------------------------------------------------------
// Kernel 2: fused WMMA GEMM (a @ th^T) + distance transform + softmax over
// each class's 10 proxies. Block = 16 rows x 80 cols (= 8 classes), 5 waves,
// one 16x16 tile per wave, K=64 -> 2x v_wmma_f32_16x16x32_f16.
// ---------------------------------------------------------------------------
__global__ __launch_bounds__(160) void podnet_fused_kernel(
    const _Float16* __restrict__ aH, const _Float16* __restrict__ tH,
    const float* __restrict__ na2, const float* __restrict__ nt2,
    float* __restrict__ out)
{
    __shared__ float dots[16 * 80];

    const int lane = threadIdx.x & 31;
    const int tile = threadIdx.x >> 5;        // 0..4
    const int row0 = blockIdx.y * 16;
    const int cb   = blockIdx.x;              // 80-column block (8 classes)
    const int n    = lane & 15;
    const int hlf  = lane >> 4;

    // ---- A fragments: lane (n, hlf) holds row M=n, K = {h*8..h*8+7, 16+h*8..}
    const _Float16* ar = aH + (size_t)(row0 + n) * DFEAT;
    v8h a0lo = *(const v8h*)(ar +  0 + hlf * 8);
    v8h a0hi = *(const v8h*)(ar + 16 + hlf * 8);
    v8h a1lo = *(const v8h*)(ar + 32 + hlf * 8);
    v8h a1hi = *(const v8h*)(ar + 48 + hlf * 8);
    v16h a0 = __builtin_shufflevector(a0lo, a0hi, 0,1,2,3,4,5,6,7,8,9,10,11,12,13,14,15);
    v16h a1 = __builtin_shufflevector(a1lo, a1hi, 0,1,2,3,4,5,6,7,8,9,10,11,12,13,14,15);

    // ---- B fragments: B[k][n] = th[c0+n][k]; lane holds K = {h*8..h*8+7, 16+h*8..}
    const int c0 = cb * 80 + tile * 16;
    const _Float16* br = tH + (size_t)(c0 + n) * DFEAT + hlf * 8;
    v16h b0 = __builtin_shufflevector(*(const v8h*)(br), *(const v8h*)(br + 16), 0,1,2,3,4,5,6,7,8,9,10,11,12,13,14,15);         // K 0..31
    v16h b1 = __builtin_shufflevector(*(const v8h*)(br + 32), *(const v8h*)(br + 48), 0,1,2,3,4,5,6,7,8,9,10,11,12,13,14,15);    // K 32..63

    v8f acc = {};
    acc = __builtin_amdgcn_wmma_f32_16x16x32_f16(false, a0, false, b0,
                                                 (short)0, acc, false, false);
    acc = __builtin_amdgcn_wmma_f32_16x16x32_f16(false, a1, false, b1,
                                                 (short)0, acc, false, false);

    // D layout: VGPR v -> M = v + 8*hlf, N = lane&15
#pragma unroll
    for (int v = 0; v < 8; ++v)
        dots[(v + 8 * hlf) * 80 + tile * 16 + n] = acc[v];

    __syncthreads();

    // ---- epilogue: 128 threads, one (row m, class cc) each
    if (threadIdx.x < 128) {
        const int m  = threadIdx.x >> 3;
        const int cc = threadIdx.x & 7;
        const float na = na2[row0 + m];
        const float* ntp = nt2 + cb * 80 + cc * 10;
        const float* sp  = dots + m * 80 + cc * 10;

        float s[10];
        float mx = -3.402823466e38f;
#pragma unroll
        for (int j = 0; j < 10; ++j) {
            float d2 = na + ntp[j] - 2.0f * sp[j];
            d2 = fmaxf(d2, 0.0f);
            s[j] = -d2;
            mx = fmaxf(mx, s[j]);
        }
        float den = 0.0f, num = 0.0f;
#pragma unroll
        for (int j = 0; j < 10; ++j) {
            const float e = __expf(s[j] - mx);
            den += e;
            num += s[j] * e;
        }
        out[(size_t)(row0 + m) * CLASSES + cb * 8 + cc] = num / den;
    }
}

// ---------------------------------------------------------------------------
extern "C" void kernel_launch(void* const* d_in, const int* in_sizes, int n_in,
                              void* d_out, int out_size, void* d_ws, size_t ws_size,
                              hipStream_t stream) {
    const float* x     = (const float*)d_in[0];   // (8192, 64) f32
    const float* theta = (const float*)d_in[1];   // (64, 10, 1000) f32
    float* out         = (float*)d_out;           // (8192, 1000) f32

    char* ws = (char*)d_ws;
    _Float16* aH = (_Float16*)ws;                               // 8192*64*2  = 1048576 B
    _Float16* tH = (_Float16*)(ws + 1048576);                   // 10000*64*2 = 1280000 B
    float*   na2 = (float*)(ws + 1048576 + 1280000);            // 8192*4  = 32768 B
    float*   nt2 = (float*)(ws + 1048576 + 1280000 + 32768);    // 10000*4 = 40000 B

    // one wave per row: 8192 + 10000 = 18192 waves -> 582144 threads
    const int totalThreads = (BATCHN + THROWS) * 32;
    podnet_prep_kernel<<<(totalThreads + 255) / 256, 256, 0, stream>>>(
        x, theta, aH, tH, na2, nt2);

    dim3 grid(THROWS / 80, BATCHN / 16);   // (125, 512)
    podnet_fused_kernel<<<grid, 160, 0, stream>>>(aH, tH, na2, nt2, out);
}
